// Attention_Filter_28570122453619
// MI455X (gfx1250) — matrix operand order repaired and checked by reading, not compiled, not run on hardware
//
#include <hip/hip_runtime.h>

typedef __attribute__((ext_vector_type(16))) __bf16 v16bf;
typedef __attribute__((ext_vector_type(8)))  __bf16 v8bf;
typedef __attribute__((ext_vector_type(8)))  float  v8f;

// ---------------------------------------------------------------- helpers

__device__ __forceinline__ __bf16 f2bf(float f){
  unsigned u = __float_as_uint(f);
  u += 0x7FFFu + ((u >> 16) & 1u);           // round-to-nearest-even
  union { unsigned short s; __bf16 b; } cv;
  cv.s = (unsigned short)(u >> 16);
  return cv.b;
}

// A-fragment (16x32, row-major source, lda multiple of 16):
// lane L: row m0+(L&15); elements 0..7 = K[h*8..h*8+7], 8..15 = K[16+h*8..], h=L>>4
__device__ __forceinline__ v16bf load_a_bf(const __bf16* A, int lda, int m0, int k0, int lane){
  int m = m0 + (lane & 15);
  int h = lane >> 4;
  const __bf16* p = A + (long long)m * lda + k0 + h * 8;
  v8bf lo = *(const v8bf*)p;
  v8bf hi = *(const v8bf*)(p + 16);
  v16bf r;
#pragma unroll
  for (int i = 0; i < 8; ++i) { r[i] = lo[i]; r[i + 8] = hi[i]; }
  return r;
}

// B-fragment (32x16): source Bt is the TRANSPOSE of the B operand, row n = column n.
// lane L: col n0+(L&15); elements 0..7 = K[h*8..h*8+7], 8..15 = K[16+h*8..], h=L>>4
__device__ __forceinline__ v16bf load_b_bf(const __bf16* Bt, int ldb, int n0, int k0, int lane){
  int n = n0 + (lane & 15);
  int h = lane >> 4;
  const __bf16* p = Bt + (long long)n * ldb + k0 + h * 8;
  v8bf lo = *(const v8bf*)p;
  v8bf hi = *(const v8bf*)(p + 16);
  v16bf r;
#pragma unroll
  for (int i = 0; i < 8; ++i) { r[i] = lo[i]; r[i + 8] = hi[i]; }
  return r;
}

__device__ __forceinline__ v8f wmma_bf(v16bf a, v16bf b, v8f c){
  return __builtin_amdgcn_wmma_f32_16x16x32_bf16(false, a, false, b, (short)0, c, false, false);
}

// 1x1 tile GEMM (small cases)
__device__ __forceinline__ v8f gemm_bf(const __bf16* A, int lda, const __bf16* Bt, int ldb,
                                       int m0, int n0, int K, int lane, v8f acc){
  for (int k0 = 0; k0 < K; k0 += 32){
    v16bf a = load_a_bf(A, lda, m0, k0, lane);
    v16bf b = load_b_bf(Bt, ldb, n0, k0, lane);
    acc = wmma_bf(a, b, acc);
  }
  return acc;
}

// 2x2 register-blocked GEMM: wave computes a 32x32 macro-tile.
// 4 WMMAs per 4 fragment loads -> 2x arithmetic intensity vs 1x1.
__device__ __forceinline__ void gemm_bf_2x2(const __bf16* A, int lda, const __bf16* Bt, int ldb,
                                            int m0, int n0, int K, int lane,
                                            v8f& d00, v8f& d01, v8f& d10, v8f& d11){
  for (int k0 = 0; k0 < K; k0 += 32){
    v16bf a0 = load_a_bf(A, lda, m0,      k0, lane);
    v16bf a1 = load_a_bf(A, lda, m0 + 16, k0, lane);
    v16bf b0 = load_b_bf(Bt, ldb, n0,      k0, lane);
    v16bf b1 = load_b_bf(Bt, ldb, n0 + 16, k0, lane);
    d00 = wmma_bf(a0, b0, d00);
    d01 = wmma_bf(a0, b1, d01);
    d10 = wmma_bf(a1, b0, d10);
    d11 = wmma_bf(a1, b1, d11);
  }
}

// 2x2 with f32 A operand converted on the fly (softmaxed attn)
__device__ __forceinline__ void gemm_f32a_2x2(const float* A, int lda, const __bf16* Bt, int ldb,
                                              int m0, int n0, int K, int lane,
                                              v8f& d00, v8f& d01, v8f& d10, v8f& d11){
  int mlo = m0 + (lane & 15);
  int h = lane >> 4;
  for (int k0 = 0; k0 < K; k0 += 32){
    const float* p0 = A + (long long)mlo * lda + k0 + h * 8;
    const float* p1 = p0 + (long long)16 * lda;
    v16bf a0, a1;
#pragma unroll
    for (int i = 0; i < 8; ++i) {
      a0[i] = f2bf(p0[i]); a0[i + 8] = f2bf(p0[16 + i]);
      a1[i] = f2bf(p1[i]); a1[i + 8] = f2bf(p1[16 + i]);
    }
    v16bf b0 = load_b_bf(Bt, ldb, n0,      k0, lane);
    v16bf b1 = load_b_bf(Bt, ldb, n0 + 16, k0, lane);
    d00 = wmma_bf(a0, b0, d00);
    d01 = wmma_bf(a0, b1, d01);
    d10 = wmma_bf(a1, b0, d10);
    d11 = wmma_bf(a1, b1, d11);
  }
}

// C/D layout: element r of lane L -> row m0+8*(L>>4)+r, col n0+(L&15)
__device__ __forceinline__ void store_f32(float* D, int ldd, int m0, int n0, int lane, v8f acc, float scale){
  int n = n0 + (lane & 15);
  int h = lane >> 4;
#pragma unroll
  for (int r = 0; r < 8; ++r) D[(long long)(m0 + 8*h + r) * ldd + n] = acc[r] * scale;
}

__device__ __forceinline__ void store_bf(__bf16* D, int ldd, int m0, int n0, int lane, v8f acc){
  int n = n0 + (lane & 15);
  int h = lane >> 4;
#pragma unroll
  for (int r = 0; r < 8; ++r) D[(long long)(m0 + 8*h + r) * ldd + n] = f2bf(acc[r]);
}

__device__ __forceinline__ void store_f32_2x2(float* D, int ldd, int m0, int n0, int lane,
                                              v8f d00, v8f d01, v8f d10, v8f d11, float scale){
  store_f32(D, ldd, m0,      n0,      lane, d00, scale);
  store_f32(D, ldd, m0,      n0 + 16, lane, d01, scale);
  store_f32(D, ldd, m0 + 16, n0,      lane, d10, scale);
  store_f32(D, ldd, m0 + 16, n0 + 16, lane, d11, scale);
}

__device__ __forceinline__ void store_bf_2x2(__bf16* D, int ldd, int m0, int n0, int lane,
                                             v8f d00, v8f d01, v8f d10, v8f d11){
  store_bf(D, ldd, m0,      n0,      lane, d00);
  store_bf(D, ldd, m0,      n0 + 16, lane, d01);
  store_bf(D, ldd, m0 + 16, n0,      lane, d10);
  store_bf(D, ldd, m0 + 16, n0 + 16, lane, d11);
}

// ---------------------------------------------------------------- small kernels

__global__ __launch_bounds__(256) void k_cvt(__bf16* dst, const float* src, int n){
  int i = blockIdx.x * 256 + threadIdx.x;
  if (i < n) dst[i] = f2bf(src[i]);
}

// Wd = bf16(w_v1 - w_v2) ; Wv1n = bf16(-w_v1)
__global__ __launch_bounds__(256) void k_wcomb(__bf16* wd, __bf16* wv1n,
                                               const float* wv1, const float* wv2, int n){
  int i = blockIdx.x * 256 + threadIdx.x;
  if (i < n){ wd[i] = f2bf(wv1[i] - wv2[i]); wv1n[i] = f2bf(-wv1[i]); }
}

// spectral gate as a real 64x64 linear map (rfft*w -> irfft, norm='ortho')
__global__ __launch_bounds__(256) void k_gate(__bf16* Mb, const float* cw){
  int idx = blockIdx.x * 256 + threadIdx.x;   // 4096 entries
  int m = idx >> 6, n = idx & 63;
  float s = cw[0];                                         // Re(w0)
  s += cw[64] * (((m + n) & 1) ? -1.f : 1.f);              // Re(w32)*(-1)^(m+n)
  float base = 2.f * 3.14159265358979323846f * (float)(m - n) / 64.f;
  for (int k = 1; k < 32; ++k){
    float th = base * (float)k;
    s += 2.f * (cw[2*k] * __cosf(th) - cw[2*k+1] * __sinf(th));
  }
  Mb[idx] = f2bf(s * (1.f / 64.f));
}

__global__ __launch_bounds__(256) void k_split(__bf16* QFb, __bf16* KFb, __bf16* VFb,
                                               __bf16* VTb, const float* qkvF){
  long long i = (long long)blockIdx.x * 256 + threadIdx.x;  // < 8*1024*768
  int c = (int)(i % 768);
  long long bn = i / 768;
  int n = (int)(bn % 1024);
  int b = (int)(bn / 1024);
  const float* row = qkvF + bn * 2304;
  QFb[i] = f2bf(row[c]);
  KFb[i] = f2bf(row[768 + c]);
  float v = row[1536 + c];
  VFb[i] = f2bf(v);
  VTb[((long long)b * 768 + c) * 1024 + n] = f2bf(v);
}

__global__ __launch_bounds__(256) void k_symrelu(float* AH){
  long long i = (long long)blockIdx.x * 256 + threadIdx.x;  // 8M
  int b = (int)(i >> 20);
  int ii = (int)((i >> 10) & 1023);
  int jj = (int)(i & 1023);
  if (jj < ii) return;
  float* M = AH + ((long long)b << 20);
  float a = M[ii * 1024 + jj];
  float c = M[jj * 1024 + ii];
  float s = fmaxf(0.5f * (a + c), 0.f);
  M[ii * 1024 + jj] = s;
  M[jj * 1024 + ii] = s;
}

__global__ __launch_bounds__(256) void k_degree(float* DEG, const float* AH){
  __shared__ float red[256];
  long long row = blockIdx.x;                 // 8192 rows
  const float* p = AH + row * 1024;
  int t = threadIdx.x;
  float s = 0.f;
  for (int j = t; j < 1024; j += 256) s += p[j];
  red[t] = s; __syncthreads();
  for (int k = 128; k > 0; k >>= 1){ if (t < k) red[t] += red[t + k]; __syncthreads(); }
  if (t == 0){
    float d = red[0];
    DEG[row] = (d != 0.f) ? rsqrtf(d) : 0.f;
  }
}

__global__ __launch_bounds__(256) void k_norm(__bf16* Sb, const float* AH, const float* DEG){
  long long i = (long long)blockIdx.x * 256 + threadIdx.x;  // 8M
  int b = (int)(i >> 20);
  int ii = (int)((i >> 10) & 1023);
  int jj = (int)(i & 1023);
  Sb[i] = f2bf(AH[i] * DEG[b * 1024 + ii] * DEG[b * 1024 + jj]);
}

__global__ __launch_bounds__(256) void k_softmax(float* attn){
  __shared__ float red[256];
  long long row = blockIdx.x;                 // 8*12*1024 rows
  float* p = attn + row * 1024;
  int t = threadIdx.x;
  float m = -3.4e38f;
  for (int j = t; j < 1024; j += 256) m = fmaxf(m, p[j]);
  red[t] = m; __syncthreads();
  for (int k = 128; k > 0; k >>= 1){ if (t < k) red[t] = fmaxf(red[t], red[t + k]); __syncthreads(); }
  m = red[0]; __syncthreads();
  float s = 0.f;
  for (int j = t; j < 1024; j += 256){ float e = __expf(p[j] - m); p[j] = e; s += e; }
  red[t] = s; __syncthreads();
  for (int k = 128; k > 0; k >>= 1){ if (t < k) red[t] += red[t + k]; __syncthreads(); }
  float inv = 1.f / red[0];
  for (int j = t; j < 1024; j += 256) p[j] *= inv;
}

// ---------------------------------------------------------------- WMMA GEMM kernels

__global__ __launch_bounds__(256) void k_gemm_qkv(float* qkvF, const __bf16* Xb, const __bf16* Wb){
  int lane = threadIdx.x & 31;
  int tile = blockIdx.x * 8 + (threadIdx.x >> 5);   // 256 x 72 macro-tiles
  int tn = tile % 72, tm = tile / 72;
  int m0 = tm * 32, n0 = tn * 32;
  v8f d00 = {}, d01 = {}, d10 = {}, d11 = {};
  gemm_bf_2x2(Xb, 768, Wb, 768, m0, n0, 768, lane, d00, d01, d10, d11);
  store_f32_2x2(qkvF, 2304, m0, n0, lane, d00, d01, d10, d11, 1.f);
}

__global__ __launch_bounds__(256) void k_logits(float* attn, const __bf16* QFb, const __bf16* KFb){
  int lane = threadIdx.x & 31;
  int tile = blockIdx.x * 8 + (threadIdx.x >> 5);   // 8*12*32*32 macro-tiles
  int tj = tile & 31, ti = (tile >> 5) & 31;
  int h = (tile >> 10) % 12, b = tile / 12288;
  const __bf16* A  = QFb + (long long)b * 786432 + h * 64;
  const __bf16* Bt = KFb + (long long)b * 786432 + h * 64;
  int m0 = ti * 32, n0 = tj * 32;
  v8f d00 = {}, d01 = {}, d10 = {}, d11 = {};
  gemm_bf_2x2(A, 768, Bt, 768, m0, n0, 64, lane, d00, d01, d10, d11);
  float* D = attn + (long long)(b * 12 + h) * 1048576;
  store_f32_2x2(D, 1024, m0, n0, lane, d00, d01, d10, d11, 0.125f);   // hd^-0.5
}

__global__ __launch_bounds__(256) void k_gram(float* AH, const __bf16* QFb, const __bf16* KFb){
  int lane = threadIdx.x & 31;
  int tile = blockIdx.x * 8 + (threadIdx.x >> 5);   // 8*32*32 macro-tiles
  int tj = tile & 31, ti = (tile >> 5) & 31, b = tile >> 10;
  int m0 = ti * 32, n0 = tj * 32;
  v8f d00 = {}, d01 = {}, d10 = {}, d11 = {};
  gemm_bf_2x2(QFb + (long long)b * 786432, 768, KFb + (long long)b * 786432, 768,
              m0, n0, 768, lane, d00, d01, d10, d11);
  store_f32_2x2(AH + ((long long)b << 20), 1024, m0, n0, lane, d00, d01, d10, d11, 1.f);
}

__global__ __launch_bounds__(256) void k_l2(__bf16* L2b, const __bf16* Sb){
  int lane = threadIdx.x & 31;
  int tile = blockIdx.x * 8 + (threadIdx.x >> 5);   // 8*32*32 macro-tiles
  int tj = tile & 31, ti = (tile >> 5) & 31, b = tile >> 10;
  const __bf16* S = Sb + ((long long)b << 20);      // symmetric: its own transpose
  int m0 = ti * 32, n0 = tj * 32;
  v8f d00 = {}, d01 = {}, d10 = {}, d11 = {};
  gemm_bf_2x2(S, 1024, S, 1024, m0, n0, 1024, lane, d00, d01, d10, d11);
  store_bf_2x2(L2b + ((long long)b << 20), 1024, m0, n0, lane, d00, d01, d10, d11);
}

__global__ __launch_bounds__(256) void k_t2(__bf16* T2b, const __bf16* L2b, const __bf16* VTb){
  int lane = threadIdx.x & 31;
  int tile = blockIdx.x * 8 + (threadIdx.x >> 5);   // 8*32*24 macro-tiles
  int tn = tile % 24, ti = (tile / 24) % 32, b = tile / 768;
  int m0 = ti * 32, n0 = tn * 32;
  v8f d00 = {}, d01 = {}, d10 = {}, d11 = {};
  gemm_bf_2x2(L2b + ((long long)b << 20), 1024, VTb + (long long)b * 786432, 1024,
              m0, n0, 1024, lane, d00, d01, d10, d11);
  store_bf_2x2(T2b + (long long)b * 786432, 768, m0, n0, lane, d00, d01, d10, d11);
}

// ifftT[b,h,d,n] = (v_head @ M^T)[n,d]   (tiny K, keep 1x1)
__global__ __launch_bounds__(256) void k_spectral(__bf16* ifftT, const __bf16* VFb, const __bf16* Mb){
  int lane = threadIdx.x & 31;
  int tile = blockIdx.x * 8 + (threadIdx.x >> 5);   // 8*12*64*4 tiles
  int tj = tile & 3, ti = (tile >> 2) & 63;
  int h = (tile >> 8) % 12, b = tile / 12288;
  v8f acc = {};
  acc = gemm_bf(VFb + (long long)b * 786432 + h * 64, 768, Mb, 64,
                ti * 16, tj * 16, 64, lane, acc);
  int d = tj * 16 + (lane & 15);
  long long base = ((long long)(b * 12 + h) * 64 + d) * 1024 + ti * 16 + 8 * (lane >> 4);
#pragma unroll
  for (int r = 0; r < 8; ++r) ifftT[base + r] = f2bf(acc[r]);
}

// YFb[b,n,h*64+d] = (attn @ ifft)[n,d]
__global__ __launch_bounds__(256) void k_y(__bf16* YFb, const float* attn, const __bf16* ifftT){
  int lane = threadIdx.x & 31;
  int tile = blockIdx.x * 8 + (threadIdx.x >> 5);   // 8*12*32*2 macro-tiles
  int tj = tile & 1, ti = (tile >> 1) & 31;
  int h = (tile >> 6) % 12, b = tile / 768;
  int m0 = ti * 32, n0 = tj * 32;
  v8f d00 = {}, d01 = {}, d10 = {}, d11 = {};
  gemm_f32a_2x2(attn + (long long)(b * 12 + h) * 1048576, 1024,
                ifftT + (long long)(b * 12 + h) * 65536, 1024,
                m0, n0, 1024, lane, d00, d01, d10, d11);
  store_bf_2x2(YFb + (long long)b * 786432 + h * 64, 768, m0, n0, lane, d00, d01, d10, d11);
}

// out0 = YF @ Wproj^T + T2 @ (Wv1-Wv2)^T + VF @ (-Wv1)^T + b_proj
__global__ __launch_bounds__(256) void k_final(float* out0, const __bf16* YFb, const __bf16* T2b,
                                               const __bf16* VFb, const __bf16* Wprojb,
                                               const __bf16* Wdb, const __bf16* Wv1nb,
                                               const float* b_proj){
  int lane = threadIdx.x & 31;
  int tile = blockIdx.x * 8 + (threadIdx.x >> 5);   // 256 x 24 macro-tiles
  int tn = tile % 24, tm = tile / 24;
  int m0 = tm * 32, n0 = tn * 32;
  v8f d00 = {}, d01 = {}, d10 = {}, d11 = {};
  gemm_bf_2x2(YFb, 768, Wprojb, 768, m0, n0, 768, lane, d00, d01, d10, d11);
  gemm_bf_2x2(T2b, 768, Wdb,    768, m0, n0, 768, lane, d00, d01, d10, d11);
  gemm_bf_2x2(VFb, 768, Wv1nb,  768, m0, n0, 768, lane, d00, d01, d10, d11);
  int nlo = n0 + (lane & 15);
  int h = lane >> 4;
  float bp0 = b_proj[nlo];
  float bp1 = b_proj[nlo + 16];
#pragma unroll
  for (int r = 0; r < 8; ++r){
    long long row0 = (long long)(m0 + 8 * h + r) * 768;
    long long row1 = (long long)(m0 + 16 + 8 * h + r) * 768;
    out0[row0 + nlo]      = d00[r] + bp0;
    out0[row0 + nlo + 16] = d01[r] + bp1;
    out0[row1 + nlo]      = d10[r] + bp0;
    out0[row1 + nlo + 16] = d11[r] + bp1;
  }
}

// ---------------------------------------------------------------- launch

extern "C" void kernel_launch(void* const* d_in, const int* in_sizes, int n_in,
                              void* d_out, int out_size, void* d_ws, size_t ws_size,
                              hipStream_t stream){
  (void)in_sizes; (void)n_in; (void)out_size; (void)ws_size;
  const float* x      = (const float*)d_in[0];
  const float* w_qkv  = (const float*)d_in[1];
  const float* w_v1   = (const float*)d_in[2];
  const float* w_v2   = (const float*)d_in[3];
  const float* w_proj = (const float*)d_in[4];
  const float* b_proj = (const float*)d_in[5];
  const float* cw     = (const float*)d_in[6];

  float* out0 = (float*)d_out;                 // [8,1024,768]
  float* attn = out0 + 6291456ll;              // [8,12,1024,1024]

  char* ws = (char*)d_ws;
  size_t off = 0;
  auto alloc = [&](size_t bytes){ void* p = ws + off; off += bytes; return p; };

  float*  qkvF   = (float*)alloc(75497472ull);               // [8192,2304] f32
  // region reused after k_split:
  float*  AH     = qkvF;                                     // [8,1024,1024] f32
  __bf16* Sb     = (__bf16*)((char*)qkvF + 33554432ull);     // [8,1024,1024] bf16
  __bf16* L2b    = (__bf16*)((char*)qkvF + 50331648ull);     // [8,1024,1024] bf16
  __bf16* Xb     = (__bf16*)alloc(12582912ull);              // [8192,768]
  __bf16* QFb    = (__bf16*)alloc(12582912ull);
  __bf16* KFb    = (__bf16*)alloc(12582912ull);
  __bf16* VFb    = (__bf16*)alloc(12582912ull);
  __bf16* VTb    = (__bf16*)alloc(12582912ull);              // [8,768,1024]
  __bf16* T2b    = (__bf16*)alloc(12582912ull);              // [8,1024,768]
  __bf16* ifftT  = (__bf16*)alloc(12582912ull);              // [8,12,64,1024]
  __bf16* YFb    = (__bf16*)alloc(12582912ull);              // [8,1024,768]
  __bf16* Wqkvb  = (__bf16*)alloc(3538944ull);               // [2304,768]
  __bf16* Wprojb = (__bf16*)alloc(1179648ull);               // [768,768]
  __bf16* Wdb    = (__bf16*)alloc(1179648ull);
  __bf16* Wv1nb  = (__bf16*)alloc(1179648ull);
  __bf16* Mb     = (__bf16*)alloc(8192ull);                  // [64,64]
  float*  DEG    = (float*)alloc(32768ull);                  // [8,1024]

  // conversions + gate matrix
  k_cvt  <<<24576, 256, 0, stream>>>(Xb, x, 6291456);
  k_cvt  <<<6912,  256, 0, stream>>>(Wqkvb, w_qkv, 1769472);
  k_cvt  <<<2304,  256, 0, stream>>>(Wprojb, w_proj, 589824);
  k_wcomb<<<2304,  256, 0, stream>>>(Wdb, Wv1nb, w_v1, w_v2, 589824);
  k_gate <<<16,    256, 0, stream>>>(Mb, cw);

  // qkv projection + operand prep
  k_gemm_qkv<<<2304,  256, 0, stream>>>(qkvF, Xb, Wqkvb);
  k_split   <<<24576, 256, 0, stream>>>(QFb, KFb, VFb, VTb, qkvF);

  // attention logits (scaled) + full-channel gram
  k_logits<<<12288, 256, 0, stream>>>(attn, QFb, KFb);
  k_gram  <<<1024,  256, 0, stream>>>(AH, QFb, KFb);

  // graph-Laplacian branch
  k_symrelu<<<32768, 256, 0, stream>>>(AH);
  k_degree <<<8192,  256, 0, stream>>>(DEG, AH);
  k_norm   <<<32768, 256, 0, stream>>>(Sb, AH, DEG);
  k_l2     <<<1024,  256, 0, stream>>>(L2b, Sb);
  k_t2     <<<768,   256, 0, stream>>>(T2b, L2b, VTb);

  // softmax + spectral gate + attention apply
  k_softmax <<<98304, 256, 0, stream>>>(attn);
  k_spectral<<<3072,  256, 0, stream>>>(ifftT, VFb, Mb);
  k_y       <<<768,   256, 0, stream>>>(YFb, attn, ifftT);

  // fused final projection + laplacian contribution + bias
  k_final<<<768, 256, 0, stream>>>(out0, YFb, T2b, VFb, Wprojb, Wdb, Wv1nb, b_proj);
}
